// GraphAttention_37409165149046
// MI455X (gfx1250) — matrix operand order repaired and checked by reading, not compiled, not run on hardware
//
#include <hip/hip_runtime.h>
#include <hip/hip_bf16.h>

#define HIDDEN   768
#define HEADS    12
#define HEAD_DIM 64
#define SEQ      1024
#define BATCH    8
#define NROWS    (BATCH * SEQ)   // 8192

typedef __attribute__((ext_vector_type(16))) __bf16 bf16x16;
typedef __attribute__((ext_vector_type(8)))  float  f32x8;
typedef __attribute__((ext_vector_type(4)))  unsigned int u32x4;
typedef __attribute__((ext_vector_type(8)))  int    i32x8;
typedef __attribute__((ext_vector_type(4)))  int    i32x4;

union Frag {
  bf16x16 v;
  uint4   q[2];
  unsigned short u[16];
};

__device__ __forceinline__ unsigned short f2bf(float f) {
  union { __bf16 h; unsigned short u; } c;
  c.h = (__bf16)f;
  return c.u;
}
// Single-instruction packed convert: dst[15:0]=bf16(lo), dst[31:16]=bf16(hi)
__device__ __forceinline__ unsigned cvt_pk(float lo, float hi) {
  unsigned r;
  asm("v_cvt_pk_bf16_f32 %0, %1, %2" : "=v"(r) : "v"(lo), "v"(hi));
  return r;
}
__device__ __forceinline__ uint4 pack8(float4 a, float4 b) {
  uint4 v;
  v.x = cvt_pk(a.x, a.y); v.y = cvt_pk(a.z, a.w);
  v.z = cvt_pk(b.x, b.y); v.w = cvt_pk(b.z, b.w);
  return v;
}
__device__ __forceinline__ f32x8 wmma_bf16(const Frag& a, const Frag& b, f32x8 c) {
  return __builtin_amdgcn_wmma_f32_16x16x32_bf16(false, a.v, false, b.v,
                                                 (short)0, c, false, false);
}

// ---------------------------------------------------------------------------
// Tensor Data Mover: 2-D bf16 tile load, Global -> LDS (D# per ISA 8.3/8.4).
// ---------------------------------------------------------------------------
#if __has_builtin(__builtin_amdgcn_tensor_load_to_lds)
#define HAVE_TDM 1
#pragma message("CDNA5: tensor_load_to_lds path ENABLED")
__device__ __forceinline__ void tdm_load_2d_bf16(unsigned lds_off, const void* gptr,
                                                 unsigned tile_d0, unsigned tile_d1,
                                                 unsigned stride0) {
  const unsigned long long ga = (unsigned long long)(uintptr_t)gptr;
  u32x4 g0;
  g0[0] = 1u;                                           // count=1, user mode
  g0[1] = lds_off;                                      // lds_addr (bytes)
  g0[2] = (unsigned)(ga & 0xFFFFFFFFu);                 // global_addr[31:0]
  g0[3] = (unsigned)((ga >> 32) & 0x1FFFFFFu) | (2u << 30);  // addr[56:32] | type=2
  i32x8 g1;
  g1[0] = (int)(1u << 16);                              // data_size=1 -> 2 bytes
  g1[1] = (int)((tile_d0 & 0xFFFFu) << 16);             // tensor_dim0[15:0]
  g1[2] = (int)((tile_d1 & 0xFFFFu) << 16);             // tensor_dim1[15:0]
  g1[3] = (int)((tile_d0 & 0xFFFFu) << 16);             // tile_dim0
  g1[4] = (int)(tile_d1 & 0xFFFFu);                     // tile_dim1 | tile_dim2=0
  g1[5] = (int)stride0;                                 // tensor_dim0_stride[31:0]
  g1[6] = 0;
  g1[7] = 0;
  const i32x4 z4 = {0, 0, 0, 0};
#if __clang_major__ >= 23
  const i32x8 z8 = {0, 0, 0, 0, 0, 0, 0, 0};
  __builtin_amdgcn_tensor_load_to_lds(g0, g1, z4, z4, z8, 0);
#else
  __builtin_amdgcn_tensor_load_to_lds(g0, g1, z4, z4, 0);
#endif
}
#else
#define HAVE_TDM 0
#pragma message("CDNA5: tensor_load_to_lds builtin NOT FOUND - cooperative fallback")
#endif

// ---------------------------------------------------------------------------
// Kernel 1: Y = X @ W^T + bias  (fp32 in, bf16 out in [B,HEADS,S,HEAD_DIM])
// Block tile 64 rows x 128 cols; 8 waves = 2 row-tiles x 4 col-tiles;
// each wave owns a 32x32 output (2 A-frags, 2 B-frags, 4 WMMAs per K-step).
// ---------------------------------------------------------------------------
__global__ __launch_bounds__(256)
void qkv_gemm(const float* __restrict__ X, const float* __restrict__ W,
              const float* __restrict__ bias, unsigned short* __restrict__ out) {
  __shared__ __align__(16) unsigned short sA[64 * 32];    // 4 KB
  __shared__ __align__(16) unsigned short sB[128 * 32];   // 8 KB

  const int t    = threadIdx.x;
  const int w    = t >> 5;
  const int lane = t & 31;
  const int hi   = (lane >> 4) & 1;
  const int l16  = lane & 15;
  const int wr   = w & 1;       // row tile of 32
  const int wc   = w >> 1;      // col tile of 32 (0..3)
  const int rowbase = blockIdx.y * 64;
  const int colbase = blockIdx.x * 128;

  f32x8 c[2][2];
  for (int ct = 0; ct < 2; ++ct) {
    const float bb = bias[colbase + wc * 32 + ct * 16 + l16];
    for (int i = 0; i < 8; ++i) { c[0][ct][i] = bb; c[1][ct][i] = bb; }
  }

  // cooperative-load indices
  const int la_r = t >> 2;          // 0..63   (A rows)
  const int la_k = (t & 3) * 8;     // 0,8,16,24
  const int lb_n = t >> 1;          // 0..127  (B cols == W rows)
  const int lb_k = (t & 1) * 16;    // 0,16
  const float* pX = X + (size_t)(rowbase + la_r) * HIDDEN + la_k;
  const float* pW = W + (size_t)(colbase + lb_n) * HIDDEN + lb_k;

  for (int ks = 0; ks < HIDDEN; ks += 32) {
    __syncthreads();
    {
      float4 f0 = *(const float4*)(pX + ks);
      float4 f1 = *(const float4*)(pX + ks + 4);
      *(uint4*)&sA[la_r * 32 + la_k] = pack8(f0, f1);
      float4 g0 = *(const float4*)(pW + ks);
      float4 g1 = *(const float4*)(pW + ks + 4);
      float4 g2 = *(const float4*)(pW + ks + 8);
      float4 g3 = *(const float4*)(pW + ks + 12);
      *(uint4*)&sB[lb_n * 32 + lb_k]     = pack8(g0, g1);
      *(uint4*)&sB[lb_n * 32 + lb_k + 8] = pack8(g2, g3);
    }
    if (ks + 32 < HIDDEN) {
      __builtin_prefetch(pX + ks + 32, 0, 3);
      __builtin_prefetch(pW + ks + 32, 0, 3);
    }
    __syncthreads();

    Frag a[2], bf[2];
    for (int rt = 0; rt < 2; ++rt) {
      const unsigned short* pa = &sA[(wr * 32 + rt * 16 + l16) * 32];
      a[rt].q[0] = *(const uint4*)(pa + hi * 16);
      a[rt].q[1] = *(const uint4*)(pa + hi * 16 + 8);
    }
    for (int ct = 0; ct < 2; ++ct) {
      const unsigned short* pb = &sB[(wc * 32 + ct * 16 + l16) * 32 + hi * 16];
      bf[ct].q[0] = *(const uint4*)pb;
      bf[ct].q[1] = *(const uint4*)(pb + 8);
    }
    for (int rt = 0; rt < 2; ++rt)
      for (int ct = 0; ct < 2; ++ct)
        c[rt][ct] = wmma_bf16(a[rt], bf[ct], c[rt][ct]);
  }

  for (int rt = 0; rt < 2; ++rt)
    for (int ct = 0; ct < 2; ++ct)
      for (int r = 0; r < 8; ++r) {
        const int row = rowbase + wr * 32 + rt * 16 + r + hi * 8;
        const int col = colbase + wc * 32 + ct * 16 + l16;
        const int b   = row >> 10;
        const int s   = row & (SEQ - 1);
        const int h   = col >> 6;
        const int d   = col & 63;
        out[(((size_t)b * HEADS + h) * SEQ + s) * HEAD_DIM + d] = f2bf(c[rt][ct][r]);
      }
}

// ---------------------------------------------------------------------------
// Kernel 2: masked flash attention. One wave per 16-query tile; 8 waves share
// K (TDM -> LDS row-major) and V (cooperative transposed store).
// ---------------------------------------------------------------------------
__global__ __launch_bounds__(256)
void attn(const unsigned short* __restrict__ Q, const unsigned short* __restrict__ K,
          const unsigned short* __restrict__ V, const int* __restrict__ adj,
          unsigned short* __restrict__ ctx) {
  __shared__ __align__(16) unsigned short sK[32 * 64];   // [key][d]
  __shared__ __align__(16) unsigned short sVT[64 * 32];  // [d][key]
  __shared__ __align__(16) unsigned short sP[8][16 * 32];

  const int t    = threadIdx.x;
  const int w    = t >> 5;
  const int lane = t & 31;
  const int hi   = (lane >> 4) & 1;
  const int l16  = lane & 15;
  const int b    = blockIdx.z;
  const int h    = blockIdx.y;
  const int q0   = blockIdx.x * 128 + w * 16;
  const size_t bh = ((size_t)b * HEADS + h) * SEQ * HEAD_DIM;
#if HAVE_TDM
  const unsigned sK_off = (unsigned)(uintptr_t)(void*)sK;  // flat lo32 == LDS byte addr
#endif

  Frag aQ0, aQ1;
  {
    const unsigned short* qp = Q + bh + (size_t)(q0 + l16) * HEAD_DIM;
    aQ0.q[0] = *(const uint4*)(qp + hi * 16);
    aQ0.q[1] = *(const uint4*)(qp + hi * 16 + 8);
    aQ1.q[0] = *(const uint4*)(qp + 32 + hi * 16);
    aQ1.q[1] = *(const uint4*)(qp + 32 + hi * 16 + 8);
  }

  float m[8], l[8];
  f32x8 o[4];
  for (int r = 0; r < 8; ++r) { m[r] = -1e30f; l[r] = 0.f; }
  for (int j = 0; j < 4; ++j)
    for (int r = 0; r < 8; ++r) o[j][r] = 0.f;

  const int lkey = t >> 3;        // 0..31
  const int ld0  = (t & 7) * 8;   // 0..56
  const int* adjrow = adj + ((size_t)b * SEQ + q0) * SEQ;
  const float scale = 0.125f;

  for (int kc = 0; kc < SEQ; kc += 32) {
    __syncthreads();
    {   // V chunk: cooperative transposed store into LDS
      union { uint4 q; unsigned short u[8]; } vv;
      vv.q = *(const uint4*)(V + bh + (size_t)(kc + lkey) * HEAD_DIM + ld0);
      for (int i = 0; i < 8; ++i) sVT[(ld0 + i) * 32 + lkey] = vv.u[i];
    }
#if HAVE_TDM
    if (w == 0) {   // K chunk: one TDM descriptor moves 32x64 bf16 tile
      tdm_load_2d_bf16(sK_off, K + bh + (size_t)kc * HEAD_DIM, 64, 32, 64);
      __builtin_amdgcn_s_wait_tensorcnt(0);
    }
#else
    {
      uint4 kv = *(const uint4*)(K + bh + (size_t)(kc + lkey) * HEAD_DIM + ld0);
      *(uint4*)&sK[lkey * 64 + ld0] = kv;
    }
#endif
    __syncthreads();

    f32x8 s0, s1;
    for (int i = 0; i < 8; ++i) { s0[i] = 0.f; s1[i] = 0.f; }
    {
      Frag bk;
      const unsigned short* kp0 = &sK[l16 * 64];
      bk.q[0] = *(const uint4*)(kp0 + hi * 16);
      bk.q[1] = *(const uint4*)(kp0 + hi * 16 + 8);
      s0 = wmma_bf16(aQ0, bk, s0);
      bk.q[0] = *(const uint4*)(kp0 + 32 + hi * 16);
      bk.q[1] = *(const uint4*)(kp0 + 32 + hi * 16 + 8);
      s0 = wmma_bf16(aQ1, bk, s0);
      const unsigned short* kp1 = &sK[(16 + l16) * 64];
      bk.q[0] = *(const uint4*)(kp1 + hi * 16);
      bk.q[1] = *(const uint4*)(kp1 + hi * 16 + 8);
      s1 = wmma_bf16(aQ0, bk, s1);
      bk.q[0] = *(const uint4*)(kp1 + 32 + hi * 16);
      bk.q[1] = *(const uint4*)(kp1 + 32 + hi * 16 + 8);
      s1 = wmma_bf16(aQ1, bk, s1);
    }

    for (int r = 0; r < 8; ++r) {
      const int rl = r + hi * 8;
      const int a0 = adjrow[(size_t)rl * SEQ + kc + l16];
      const int a1 = adjrow[(size_t)rl * SEQ + kc + 16 + l16];
      float v0 = (a0 == 0) ? -1e9f : s0[r] * scale;
      float v1 = (a1 == 0) ? -1e9f : s1[r] * scale;
      float mx = fmaxf(v0, v1);
      for (int off = 1; off < 16; off <<= 1)
        mx = fmaxf(mx, __shfl_xor(mx, off, 16));
      const float mt    = fmaxf(m[r], mx);
      const float alpha = __expf(m[r] - mt);
      const float e0 = __expf(v0 - mt);
      const float e1 = __expf(v1 - mt);
      float ls = e0 + e1;
      for (int off = 1; off < 16; off <<= 1)
        ls += __shfl_xor(ls, off, 16);
      l[r] = l[r] * alpha + ls;
      m[r] = mt;
      for (int j = 0; j < 4; ++j) o[j][r] *= alpha;
      sP[w][rl * 32 + l16]      = f2bf(e0);
      sP[w][rl * 32 + 16 + l16] = f2bf(e1);
    }
    asm volatile("s_wait_dscnt 0" ::: "memory");   // same-wave LDS RAW guard

    Frag ap;
    const unsigned short* pp = &sP[w][0];
    ap.q[0] = *(const uint4*)(pp + l16 * 32 + hi * 16);
    ap.q[1] = *(const uint4*)(pp + l16 * 32 + hi * 16 + 8);
    for (int j = 0; j < 4; ++j) {
      Frag bv;
      const unsigned short* vp = &sVT[(j * 16 + l16) * 32 + hi * 16];
      bv.q[0] = *(const uint4*)vp;
      bv.q[1] = *(const uint4*)(vp + 8);
      o[j] = wmma_bf16(ap, bv, o[j]);
    }
  }

  for (int r = 0; r < 8; ++r) {
    const float inv = 1.0f / l[r];
    const int row = q0 + r + hi * 8;
    unsigned short* cp =
        ctx + ((size_t)b * SEQ + row) * HIDDEN + h * HEAD_DIM;
    for (int j = 0; j < 4; ++j) cp[j * 16 + l16] = f2bf(o[j][r] * inv);
  }
}

// ---------------------------------------------------------------------------
// Kernel 3: x = ctx @ Wo^T + bo + inputs  (fp32 out; ctx tile via TDM)
// Same 64x128 block tiling as qkv_gemm.
// ---------------------------------------------------------------------------
__global__ __launch_bounds__(256)
void out_proj(const unsigned short* __restrict__ ctx, const float* __restrict__ W,
              const float* __restrict__ bias, const float* __restrict__ resid,
              float* __restrict__ xout) {
  __shared__ __align__(16) unsigned short sA[64 * 32];
  __shared__ __align__(16) unsigned short sB[128 * 32];

  const int t    = threadIdx.x;
  const int w    = t >> 5;
  const int lane = t & 31;
  const int hi   = (lane >> 4) & 1;
  const int l16  = lane & 15;
  const int wr   = w & 1;
  const int wc   = w >> 1;
  const int rowbase = blockIdx.y * 64;
  const int colbase = blockIdx.x * 128;
#if HAVE_TDM
  const unsigned sA_off = (unsigned)(uintptr_t)(void*)sA;
#endif

  f32x8 c[2][2];
  for (int ct = 0; ct < 2; ++ct) {
    const float bb = bias[colbase + wc * 32 + ct * 16 + l16];
    for (int i = 0; i < 8; ++i) { c[0][ct][i] = bb; c[1][ct][i] = bb; }
  }

  const int la_r = t >> 2;
  const int la_k = (t & 3) * 8;
  const int lb_n = t >> 1;
  const int lb_k = (t & 1) * 16;
  const float* pW = W + (size_t)(colbase + lb_n) * HIDDEN + lb_k;

  for (int ks = 0; ks < HIDDEN; ks += 32) {
    __syncthreads();
    {
      float4 g0 = *(const float4*)(pW + ks);
      float4 g1 = *(const float4*)(pW + ks + 4);
      float4 g2 = *(const float4*)(pW + ks + 8);
      float4 g3 = *(const float4*)(pW + ks + 12);
      *(uint4*)&sB[lb_n * 32 + lb_k]     = pack8(g0, g1);
      *(uint4*)&sB[lb_n * 32 + lb_k + 8] = pack8(g2, g3);
    }
#if HAVE_TDM
    if (w == 0) {   // 64-row x 32-col bf16 tile of ctx, row stride 768
      tdm_load_2d_bf16(sA_off, ctx + (size_t)rowbase * HIDDEN + ks, 32, 64, HIDDEN);
      __builtin_amdgcn_s_wait_tensorcnt(0);
    }
#else
    {
      uint4 cv = *(const uint4*)(ctx + (size_t)(rowbase + la_r) * HIDDEN + ks + la_k);
      *(uint4*)&sA[la_r * 32 + la_k] = cv;
    }
#endif
    if (ks + 32 < HIDDEN)
      __builtin_prefetch(pW + ks + 32, 0, 3);
    __syncthreads();

    Frag a[2], bf[2];
    for (int rt = 0; rt < 2; ++rt) {
      const unsigned short* pa = &sA[(wr * 32 + rt * 16 + l16) * 32];
      a[rt].q[0] = *(const uint4*)(pa + hi * 16);
      a[rt].q[1] = *(const uint4*)(pa + hi * 16 + 8);
    }
    for (int ct = 0; ct < 2; ++ct) {
      const unsigned short* pb = &sB[(wc * 32 + ct * 16 + l16) * 32 + hi * 16];
      bf[ct].q[0] = *(const uint4*)pb;
      bf[ct].q[1] = *(const uint4*)(pb + 8);
    }
    for (int rt = 0; rt < 2; ++rt)
      for (int ct = 0; ct < 2; ++ct)
        c[rt][ct] = wmma_bf16(a[rt], bf[ct], c[rt][ct]);
  }

  for (int rt = 0; rt < 2; ++rt)
    for (int ct = 0; ct < 2; ++ct)
      for (int r = 0; r < 8; ++r) {
        const int row = rowbase + wr * 32 + rt * 16 + r + hi * 8;
        const int col = colbase + wc * 32 + ct * 16 + l16;
        xout[(size_t)row * HIDDEN + col] =
            c[rt][ct][r] + resid[(size_t)row * HIDDEN + col];
      }
}

// ---------------------------------------------------------------------------
// Kernel 4: LayerNorm over H=768, one block per row.
// ---------------------------------------------------------------------------
__global__ __launch_bounds__(256)
void layernorm(const float* __restrict__ x, const float* __restrict__ gamma,
               const float* __restrict__ beta, float* __restrict__ out) {
  __shared__ float red[8];
  const int row = blockIdx.x;
  const int t = threadIdx.x;
  const float* xr = x + (size_t)row * HIDDEN;

  float v0 = xr[t], v1 = xr[t + 256], v2 = xr[t + 512];
  float s = v0 + v1 + v2;
  for (int off = 1; off < 32; off <<= 1) s += __shfl_xor(s, off, 32);
  if ((t & 31) == 0) red[t >> 5] = s;
  __syncthreads();
  float tot = 0.f;
  for (int i = 0; i < 8; ++i) tot += red[i];
  const float mean = tot * (1.0f / HIDDEN);

  const float d0 = v0 - mean, d1 = v1 - mean, d2 = v2 - mean;
  float vs = d0 * d0 + d1 * d1 + d2 * d2;
  for (int off = 1; off < 32; off <<= 1) vs += __shfl_xor(vs, off, 32);
  __syncthreads();
  if ((t & 31) == 0) red[t >> 5] = vs;
  __syncthreads();
  float vtot = 0.f;
  for (int i = 0; i < 8; ++i) vtot += red[i];
  const float rs = rsqrtf(vtot * (1.0f / HIDDEN) + 1e-5f);

  float* op = out + (size_t)row * HIDDEN;
  op[t]       = d0 * rs * gamma[t]       + beta[t];
  op[t + 256] = d1 * rs * gamma[t + 256] + beta[t + 256];
  op[t + 512] = d2 * rs * gamma[t + 512] + beta[t + 512];
}

// ---------------------------------------------------------------------------
extern "C" void kernel_launch(void* const* d_in, const int* in_sizes, int n_in,
                              void* d_out, int out_size, void* d_ws, size_t ws_size,
                              hipStream_t stream) {
  const float* inputs = (const float*)d_in[0];
  const int*   adj    = (const int*)d_in[1];
  const float* Wq = (const float*)d_in[2];
  const float* bq = (const float*)d_in[3];
  const float* Wk = (const float*)d_in[4];
  const float* bk = (const float*)d_in[5];
  const float* Wv = (const float*)d_in[6];
  const float* bv = (const float*)d_in[7];
  const float* Wo = (const float*)d_in[8];
  const float* bo = (const float*)d_in[9];
  const float* gamma = (const float*)d_in[10];
  const float* beta  = (const float*)d_in[11];

  const size_t QKV_ELEMS = (size_t)BATCH * HEADS * SEQ * HEAD_DIM;  // 6291456
  unsigned short* qws   = (unsigned short*)d_ws;
  unsigned short* kws   = qws + QKV_ELEMS;
  unsigned short* vws   = kws + QKV_ELEMS;
  unsigned short* ctxws = vws + QKV_ELEMS;
  float*          xbuf  = (float*)(ctxws + QKV_ELEMS);

  dim3 gG(HIDDEN / 128, NROWS / 64);   // (6, 128)
  qkv_gemm<<<gG, 256, 0, stream>>>(inputs, Wq, bq, qws);
  qkv_gemm<<<gG, 256, 0, stream>>>(inputs, Wk, bk, kws);
  qkv_gemm<<<gG, 256, 0, stream>>>(inputs, Wv, bv, vws);

  attn<<<dim3(SEQ / 128, HEADS, BATCH), 256, 0, stream>>>(qws, kws, vws, adj, ctxws);

  out_proj<<<gG, 256, 0, stream>>>(ctxws, Wo, bo, inputs, xbuf);

  layernorm<<<NROWS, 256, 0, stream>>>(xbuf, gamma, beta, (float*)d_out);
}
